// SoftTripletLoss_15796889714897
// MI455X (gfx1250) — matrix operand order repaired and checked by reading, not compiled, not run on hardware
//
#include <hip/hip_runtime.h>

typedef __attribute__((ext_vector_type(16))) _Float16 v16h;
typedef __attribute__((ext_vector_type(8)))  _Float16 v8h;
typedef __attribute__((ext_vector_type(8)))  float    v8f;

#define DIMS 512   // feature dimension D (multiple of 32 required by K-chunking)

// ---------------------------------------------------------------------------
// Kernel 1: one wave (32 lanes) per row.
//   - convert row of x (f32) to f16 into workspace (WMMA operand precision)
//   - compute xn[row] = sum_j x[row][j]^2 via wave reduction
// ---------------------------------------------------------------------------
__global__ void prep_kernel(const float* __restrict__ x,
                            _Float16* __restrict__ xh,
                            float* __restrict__ xnorm, int N) {
  int wave = (int)((blockIdx.x * blockDim.x + threadIdx.x) >> 5);
  int lane = (int)(threadIdx.x & 31);
  if (wave >= N) return;                      // wave-uniform exit

  const float* row  = x  + (size_t)wave * DIMS;
  _Float16*    hrow = xh + (size_t)wave * DIMS;

  int base = lane * 16;                       // 16 floats per lane, 64B aligned
  float vals[16];
#pragma unroll
  for (int i = 0; i < 4; ++i)
    *(float4*)&vals[i * 4] = ((const float4*)(row + base))[i];

  float ss = 0.f;
  v8h h0, h1;
#pragma unroll
  for (int i = 0; i < 8; ++i) {
    ss += vals[i] * vals[i];
    ss += vals[i + 8] * vals[i + 8];
    h0[i] = (_Float16)vals[i];
    h1[i] = (_Float16)vals[i + 8];
  }
  // 16 halves = 32 bytes per lane, two b128 stores
  ((v8h*)(hrow + base))[0] = h0;
  ((v8h*)(hrow + base))[1] = h1;

#pragma unroll
  for (int off = 16; off; off >>= 1) ss += __shfl_xor(ss, off, 32);
  if (lane == 0) xnorm[wave] = ss;
}

// ---------------------------------------------------------------------------
// Kernel 2: one wave per tile of 16 triplets.
//   A tile = 16 anchor rows, B = 16 pos rows (as columns), C = A x B^T via
//   v_wmma_f32_16x16x32_f16 accumulated over K=512; same for neg rows.
//   Diagonal of each C gives the 16 needed dot products.
// ---------------------------------------------------------------------------
__global__ void triplet_kernel(const _Float16* __restrict__ xh,
                               const float* __restrict__ xnorm,
                               const int* __restrict__ trip,
                               float* __restrict__ partials,
                               int T, int numTiles) {
  __shared__ float lsum[8];
  int lane = (int)(threadIdx.x & 31);
  int wib  = (int)(threadIdx.x >> 5);
  int tile = (int)blockIdx.x * 8 + wib;

  float contrib = 0.f;
  int   ai = 0, pi = 0, ni = 0;
  float dotP = 0.f, dotN = 0.f;
  bool  active = (tile < numTiles);           // wave-uniform

  if (active) {
    int t0 = tile * 16;
    int li = lane & 15;
    int hi = lane >> 4;
    int t  = t0 + li; if (t >= T) t = T - 1;  // lane-level clamp (no branch)
    ai = trip[3 * t + 0];
    pi = trip[3 * t + 1];
    ni = trip[3 * t + 2];
    const _Float16* rowA = xh + (size_t)ai * DIMS;
    const _Float16* rowP = xh + (size_t)pi * DIMS;
    const _Float16* rowN = xh + (size_t)ni * DIMS;

    v8f cp = {}; v8f cn = {};
#pragma unroll 4
    for (int kc = 0; kc < DIMS; kc += 32) {
      // A fragment: lane holds 16 contiguous K values at kc + hi*16
      int aoff = kc + hi * 16;
      v8h a_lo = *(const v8h*)(rowA + aoff);
      v8h a_hi = *(const v8h*)(rowA + aoff + 8);
      v16h afrag = __builtin_shufflevector(a_lo, a_hi,
          0,1,2,3,4,5,6,7,8,9,10,11,12,13,14,15);
      // B fragments: lane holds 16 contiguous K values at kc + hi*16
      int boff = kc + hi * 16;
      v8h bp_lo = *(const v8h*)(rowP + boff);
      v8h bp_hi = *(const v8h*)(rowP + boff + 8);
      v16h bpfrag = __builtin_shufflevector(bp_lo, bp_hi,
          0,1,2,3,4,5,6,7,8,9,10,11,12,13,14,15);
      v8h bn_lo = *(const v8h*)(rowN + boff);
      v8h bn_hi = *(const v8h*)(rowN + boff + 8);
      v16h bnfrag = __builtin_shufflevector(bn_lo, bn_hi,
          0,1,2,3,4,5,6,7,8,9,10,11,12,13,14,15);

      cp = __builtin_amdgcn_wmma_f32_16x16x32_f16(
              false, afrag, false, bpfrag, (short)0, cp, false, false);
      cn = __builtin_amdgcn_wmma_f32_16x16x32_f16(
              false, afrag, false, bnfrag, (short)0, cn, false, false);
    }

    // Diagonal extraction: C[i][i] -> lane i  (i<8: vgpr i lane i; i>=8: vgpr i-8 lane i+16)
#pragma unroll
    for (int r = 0; r < 8; ++r) {
      float pLo = __shfl(cp[r], r,      32);
      float pHi = __shfl(cp[r], r + 24, 32);
      float nLo = __shfl(cn[r], r,      32);
      float nHi = __shfl(cn[r], r + 24, 32);
      if (lane == r)     { dotP = pLo; dotN = nLo; }
      if (lane == r + 8) { dotP = pHi; dotN = nHi; }
    }

    if (lane < 16 && (t0 + lane) < T) {
      float xa = xnorm[ai], xp = xnorm[pi], xq = xnorm[ni];
      float dap  = fmaxf(xa + xp - 2.0f * dotP, 0.0f);
      float dan  = fmaxf(xa + xq - 2.0f * dotN, 0.0f);
      float diff = dap - dan;
      // numerically stable softplus
      contrib = fmaxf(diff, 0.0f) + log1pf(expf(-fabsf(diff)));
    }
  }

  // wave reduction, then deterministic block reduction
#pragma unroll
  for (int off = 16; off; off >>= 1) contrib += __shfl_xor(contrib, off, 32);
  if (lane == 0) lsum[wib] = contrib;
  __syncthreads();
  if (threadIdx.x == 0) {
    float s = 0.f;
#pragma unroll
    for (int w = 0; w < 8; ++w) s += lsum[w];
    partials[blockIdx.x] = s;
  }
}

// ---------------------------------------------------------------------------
// Kernel 3: deterministic final reduction, write mean.
// ---------------------------------------------------------------------------
__global__ void reduce_kernel(const float* __restrict__ partials, int n,
                              float invT, float* __restrict__ out) {
  __shared__ float s[256];
  float acc = 0.f;
  for (int i = (int)threadIdx.x; i < n; i += 256) acc += partials[i];
  s[threadIdx.x] = acc;
  __syncthreads();
  for (int off = 128; off; off >>= 1) {
    if ((int)threadIdx.x < off) s[threadIdx.x] += s[threadIdx.x + off];
    __syncthreads();
  }
  if (threadIdx.x == 0) out[0] = s[0] * invT;
}

extern "C" void kernel_launch(void* const* d_in, const int* in_sizes, int n_in,
                              void* d_out, int out_size, void* d_ws, size_t ws_size,
                              hipStream_t stream) {
  const float* x    = (const float*)d_in[0];
  const int*   trip = (const int*)d_in[1];   // JAX default: int32 (x64 disabled)
  int N = in_sizes[0] / DIMS;
  int T = in_sizes[1] / 3;

  // workspace layout: [ x as f16 | xnorm f32 | block partials f32 ]
  _Float16* xh     = (_Float16*)d_ws;
  float*    xnorm  = (float*)((char*)d_ws + (size_t)N * DIMS * sizeof(_Float16));
  float*    parts  = xnorm + N;

  int prepBlocks = (N + 7) / 8;              // 8 waves/block, 1 row/wave
  prep_kernel<<<prepBlocks, 256, 0, stream>>>(x, xh, xnorm, N);

  int numTiles = (T + 15) / 16;
  int tBlocks  = (numTiles + 7) / 8;         // 8 waves/block, 1 tile/wave
  triplet_kernel<<<tBlocks, 256, 0, stream>>>(xh, xnorm, trip, parts, T, numTiles);

  reduce_kernel<<<1, 256, 0, stream>>>(parts, tBlocks, 1.0f / (float)T,
                                       (float*)d_out);
}
